// Net_20461224198440
// MI455X (gfx1250) — matrix operand order repaired and checked by reading, not compiled, not run on hardware
//
#include <hip/hip_runtime.h>

typedef __attribute__((ext_vector_type(16))) _Float16 v16h;
typedef __attribute__((ext_vector_type(8)))  float    v8f;

#define BM 256
#define BN 128
#define BK 32
#define LDK 40   // padded LDS row stride (elements) -> 80B, kills bank conflicts, keeps 16B align

// ---------------- conversion kernels ----------------

// activation f32 [M,K] -> f16 [M,Kpad] (zero-pad K)
__global__ void cvt_act(const float* __restrict__ in, _Float16* __restrict__ out,
                        int M, int K, int Kpad) {
  long i = (long)blockIdx.x * blockDim.x + threadIdx.x;
  long total = (long)M * Kpad;
  if (i >= total) return;
  int k = (int)(i % Kpad);
  long m = i / Kpad;
  float v = (k < K) ? in[m * K + k] : 0.0f;
  out[i] = (_Float16)v;
}

// weight f32 [K,N] -> transposed f16 [Npad,Kpad] (zero-pad both)
__global__ void cvt_wt(const float* __restrict__ W, _Float16* __restrict__ Wt,
                       int K, int N, int Kpad, int Npad) {
  long i = (long)blockIdx.x * blockDim.x + threadIdx.x;
  long total = (long)Npad * Kpad;
  if (i >= total) return;
  int k = (int)(i % Kpad);
  int n = (int)(i / Kpad);
  float v = (k < K && n < N) ? W[(long)k * N + n] : 0.0f;
  Wt[i] = (_Float16)v;
}

// ---------------- WMMA GEMM: C = relu(A @ Bt^T + bias) ----------------
// A:  f16 [M, KPAD] row-major
// Bt: f16 [Npad, KPAD] row-major (B transposed, K contiguous)
// C:  OutT [M, NOUT], only cols < Nreal written
// 256x128 block tile, 8 waves, each wave owns a 64x64 sub-tile (4x4 fragments
// -> 16 WMMAs per K-step from 8 fragment loads: 1.0 ds_load_b128 per WMMA).
// Double-buffered LDS, one barrier per K-step; global loads for tile k+1
// overlap the WMMAs of tile k; prefetch warms L2 two tiles ahead.
template <typename OutT, int NOUT, int KPAD>
__global__ __launch_bounds__(256)
void gemm_wmma_f16(const _Float16* __restrict__ A, const _Float16* __restrict__ Bt,
                   const float* __restrict__ bias, OutT* __restrict__ C,
                   int Nreal) {
  __shared__ _Float16 As[2][BM * LDK];
  __shared__ _Float16 Bs[2][BN * LDK];

  const int tid   = threadIdx.x;
  const int lane  = tid & 31;
  const int wave  = tid >> 5;
  const int hlf   = lane >> 4;    // lane half: 0 or 1
  const int l16   = lane & 15;
  const int waveM = wave & 3;     // 4 wave-rows  -> 64 rows each
  const int waveN = wave >> 2;    // 2 wave-cols  -> 64 cols each

  const int m0 = blockIdx.y * BM;
  const int n0 = blockIdx.x * BN;

  v8f acc[4][4];
#pragma unroll
  for (int i = 0; i < 4; ++i)
#pragma unroll
    for (int j = 0; j < 4; ++j) acc[i][j] = (v8f)0.0f;

  // staging:
  //  A tile: 256 rows x 32 halves; thread tid copies one full row (64B = 4x uint4)
  //  B tile: 128 rows; thread copies half a row (32B = 2x uint4)
  const int brow = tid >> 1;
  const int bcol = (tid & 1) * 16;

  const _Float16* gAptr = A  + (size_t)(m0 + tid) * KPAD;
  const _Float16* gBptr = Bt + (size_t)(n0 + brow) * KPAD + bcol;

  // preload tile 0 straight into buffer 0
  {
    const uint4* gA = reinterpret_cast<const uint4*>(gAptr);
    const uint4* gB = reinterpret_cast<const uint4*>(gBptr);
    uint4* sA = reinterpret_cast<uint4*>(&As[0][tid * LDK]);
    uint4* sB = reinterpret_cast<uint4*>(&Bs[0][brow * LDK + bcol]);
    sA[0] = gA[0]; sA[1] = gA[1]; sA[2] = gA[2]; sA[3] = gA[3];
    sB[0] = gB[0]; sB[1] = gB[1];
  }

  int cur = 0;
  for (int k0 = 0; k0 < KPAD; k0 += BK) {
    __syncthreads();

    // fetch next tile into registers while this tile computes
    uint4 ra0, ra1, ra2, ra3, rb0, rb1;
    const bool more = (k0 + BK) < KPAD;
    if (more) {
      const uint4* gA = reinterpret_cast<const uint4*>(gAptr + (k0 + BK));
      const uint4* gB = reinterpret_cast<const uint4*>(gBptr + (k0 + BK));
      ra0 = gA[0]; ra1 = gA[1]; ra2 = gA[2]; ra3 = gA[3];
      rb0 = gB[0]; rb1 = gB[1];
      if (k0 + 2 * BK < KPAD) {  // warm L2 two tiles ahead
        __builtin_prefetch(gAptr + (k0 + 2 * BK), 0, 1);
        __builtin_prefetch(gBptr + (k0 + 2 * BK), 0, 1);
      }
    }

    // A fragments (16-bit A 16x32 layout: half 0 -> K 0..7,16..23 ; half 1 -> K 8..15,24..31)
    v16h afrag[4];
#pragma unroll
    for (int fm = 0; fm < 4; ++fm) {
      const int m = waveM * 64 + fm * 16 + l16;
      const _Float16* p = &As[cur][m * LDK];
      union { uint4 u[2]; v16h v; } ua;
      ua.u[0] = *reinterpret_cast<const uint4*>(p + hlf * 8);
      ua.u[1] = *reinterpret_cast<const uint4*>(p + 16 + hlf * 8);
      afrag[fm] = ua.v;
    }
    // B fragments (32x16 B: half 0 -> K 0..7,16..23 ; half 1 -> K 8..15,24..31 of col l16)
    v16h bfrag[4];
#pragma unroll
    for (int fn = 0; fn < 4; ++fn) {
      const int n = waveN * 64 + fn * 16 + l16;
      const _Float16* p = &Bs[cur][n * LDK + hlf * 8];
      union { uint4 u[2]; v16h v; } ub;
      ub.u[0] = reinterpret_cast<const uint4*>(p)[0];
      ub.u[1] = reinterpret_cast<const uint4*>(p + 16)[0];
      bfrag[fn] = ub.v;
    }

#pragma unroll
    for (int fm = 0; fm < 4; ++fm)
#pragma unroll
      for (int fn = 0; fn < 4; ++fn)
        acc[fm][fn] = __builtin_amdgcn_wmma_f32_16x16x32_f16(
            false, afrag[fm], false, bfrag[fn],
            (short)0, acc[fm][fn], false, false);

    // park the prefetched tile in the other buffer
    if (more) {
      const int nxt = cur ^ 1;
      uint4* sA = reinterpret_cast<uint4*>(&As[nxt][tid * LDK]);
      uint4* sB = reinterpret_cast<uint4*>(&Bs[nxt][brow * LDK + bcol]);
      sA[0] = ra0; sA[1] = ra1; sA[2] = ra2; sA[3] = ra3;
      sB[0] = rb0; sB[1] = rb1;
      cur = nxt;
    }
  }

  // epilogue: C/D layout -> VGPR i holds M = i + 8*hlf, N = l16
#pragma unroll
  for (int fm = 0; fm < 4; ++fm) {
#pragma unroll
    for (int fn = 0; fn < 4; ++fn) {
      const int col = n0 + waveN * 64 + fn * 16 + l16;
      if (col >= Nreal) continue;
      const float bv = bias[col];
      const int row0 = m0 + waveM * 64 + fm * 16 + hlf * 8;
      OutT* cptr = C + (size_t)row0 * NOUT + col;
#pragma unroll
      for (int i = 0; i < 8; ++i) {
        float v = acc[fm][fn][i] + bv;
        v = v > 0.0f ? v : 0.0f;
        cptr[i * NOUT] = (OutT)v;   // constant immediate offsets
      }
    }
  }
}

// ---------------- top-k + stripe gating ----------------
// one block (256 threads) per row of 2048; keep per-row top-128 values,
// then keep top-8 of 64 stripe sums; write gated row as f16.
__global__ __launch_bounds__(256)
void gate_topk(const float* __restrict__ H, _Float16* __restrict__ Out) {
  __shared__ float sv[2048];
  __shared__ float sred[256];
  __shared__ float ssum[64];
  __shared__ float skeep[64];
  __shared__ int   scnt;

  const int r = blockIdx.x;
  const int t = threadIdx.x;
  const float* row = H + (long)r * 2048;

  float lmax = 0.0f;
  for (int i = t; i < 2048; i += 256) {
    float v = row[i];
    sv[i] = v;
    lmax = fmaxf(lmax, v);
  }
  sred[t] = lmax;
  __syncthreads();
  for (int s = 128; s > 0; s >>= 1) {
    if (t < s) sred[t] = fmaxf(sred[t], sred[t + s]);
    __syncthreads();
  }

  // binary search threshold for the 128th-largest value (all values >= 0 post-ReLU)
  float lo = 0.0f, hi = sred[0];
  for (int it = 0; it < 24; ++it) {
    const float mid = 0.5f * (lo + hi);
    int c = 0;
    for (int i = t; i < 2048; i += 256) c += (sv[i] > mid) ? 1 : 0;
    if (t == 0) scnt = 0;
    __syncthreads();
    atomicAdd(&scnt, c);
    __syncthreads();
    const int total = scnt;
    __syncthreads();
    if (total >= 128) lo = mid; else hi = mid;
  }

  // neuron gating (zeros stay zero; threshold lo keeps ~top-128)
  for (int i = t; i < 2048; i += 256) {
    const float v = sv[i];
    sv[i] = (v > lo) ? v : 0.0f;
  }
  __syncthreads();

  // stripe sums (mean up to a constant factor) + exact top-8-of-64 with index tiebreak
  if (t < 64) {
    float s = 0.0f;
#pragma unroll 4
    for (int j = 0; j < 32; ++j) s += sv[t * 32 + j];
    ssum[t] = s;
  }
  __syncthreads();
  if (t < 64) {
    const float ms = ssum[t];
    int c = 0;
    for (int j = 0; j < 64; ++j) {
      const float mj = ssum[j];
      c += (mj > ms || (mj == ms && j < t)) ? 1 : 0;
    }
    skeep[t] = (c < 8) ? 1.0f : 0.0f;
  }
  __syncthreads();

  for (int i = t; i < 2048; i += 256)
    Out[(long)r * 2048 + i] = (_Float16)(sv[i] * skeep[i >> 5]);
}

// ---------------- launch ----------------

extern "C" void kernel_launch(void* const* d_in, const int* in_sizes, int n_in,
                              void* d_out, int out_size, void* d_ws, size_t ws_size,
                              hipStream_t stream) {
  (void)in_sizes; (void)n_in; (void)out_size; (void)ws_size;

  const float* x  = (const float*)d_in[0];
  const float* W1 = (const float*)d_in[1];
  const float* b1 = (const float*)d_in[2];
  const float* W2 = (const float*)d_in[3];
  const float* b2 = (const float*)d_in[4];
  const float* W3 = (const float*)d_in[5];
  const float* b3 = (const float*)d_in[6];
  const float* W4 = (const float*)d_in[7];
  const float* b4 = (const float*)d_in[8];
  float* out = (float*)d_out;

  const int Bsz = 16384;
  const int K1 = 784, K1p = 800;   // pad 784 -> 25*32
  const int N1 = 1024;
  const int N2 = 2048;
  const int N3 = 1024;
  const int N4 = 784, N4p = 896;   // pad 784 -> 7*128

  // workspace layout (all sizes are multiples of 256 bytes)
  char* ws = (char*)d_ws;
  size_t off = 0;
  auto take = [&](size_t bytes) { char* p = ws + off; off += (bytes + 255) & ~(size_t)255; return p; };

  _Float16* xb  = (_Float16*)take((size_t)Bsz * K1p * 2);   // x  f16 padded
  _Float16* W1t = (_Float16*)take((size_t)N1  * K1p * 2);
  _Float16* W2t = (_Float16*)take((size_t)N2  * N1  * 2);
  _Float16* W3t = (_Float16*)take((size_t)N3  * N2  * 2);
  _Float16* W4t = (_Float16*)take((size_t)N4p * N3  * 2);
  _Float16* h1  = (_Float16*)take((size_t)Bsz * N1  * 2);
  float*    h2f = (float*)   take((size_t)Bsz * N2  * 4);
  _Float16* h2g = (_Float16*)take((size_t)Bsz * N2  * 2);
  _Float16* h3  = (_Float16*)take((size_t)Bsz * N3  * 2);

  const int TPB = 256;
  auto blocks = [](long n) { return (unsigned)((n + 255) / 256); };

  // conversions
  cvt_act<<<blocks((long)Bsz * K1p), TPB, 0, stream>>>(x, xb, Bsz, K1, K1p);
  cvt_wt <<<blocks((long)N1 * K1p), TPB, 0, stream>>>(W1, W1t, K1, N1, K1p, N1);
  cvt_wt <<<blocks((long)N2 * N1 ), TPB, 0, stream>>>(W2, W2t, N1, N2, N1, N2);
  cvt_wt <<<blocks((long)N3 * N2 ), TPB, 0, stream>>>(W3, W3t, N2, N3, N2, N3);
  cvt_wt <<<blocks((long)N4p* N3 ), TPB, 0, stream>>>(W4, W4t, N3, N4, N3, N4p);

  const dim3 g1(N1 / BN, Bsz / BM);
  const dim3 g2(N2 / BN, Bsz / BM);
  const dim3 g3(N3 / BN, Bsz / BM);
  const dim3 g4(N4p / BN, Bsz / BM);

  // layer 1: h1 = relu(xb @ W1 + b1)        [B,1024] f16
  gemm_wmma_f16<_Float16, 1024, 800><<<g1, TPB, 0, stream>>>(xb, W1t, b1, h1, N1);
  // layer 2: h2f = relu(h1 @ W2 + b2)       [B,2048] f32 (for gating)
  gemm_wmma_f16<float, 2048, 1024><<<g2, TPB, 0, stream>>>(h1, W2t, b2, h2f, N2);
  // gating: top-128 neurons + top-8 stripes -> h2g f16
  gate_topk<<<Bsz, TPB, 0, stream>>>(h2f, h2g);
  // layer 3: h3 = relu(h2g @ W3 + b3)       [B,1024] f16
  gemm_wmma_f16<_Float16, 1024, 2048><<<g3, TPB, 0, stream>>>(h2g, W3t, b3, h3, N3);
  // layer 4: out = relu(h3 @ W4 + b4)       [B,784] f32
  gemm_wmma_f16<float, 784, 1024><<<g4, TPB, 0, stream>>>(h3, W4t, b4, out, N4);
}
